// MoleculeGCN_24197845745884
// MI455X (gfx1250) — compile-verified, matrix operand order checked by reading, not run on hardware
//
#include <hip/hip_runtime.h>
#include <hip/hip_bf16.h>
#include <stdint.h>

#define NNODES 50000
#define NEDGES 600000
#define DIM    128
#define NGR    512
#define RDIM   64
#define LN2F   0.6931471805599453f
#define NTILES (NEDGES / 16)      // 37500
#define GEMM_BLOCKS 3750          // persistent blocks; 10 row tiles each

typedef __attribute__((ext_vector_type(16))) __bf16 v16bf;
typedef __attribute__((ext_vector_type(8)))  float  v8f;

#define ENC_NEGINF 0x007FFFFFu   // order-encoded -inf

// ---------------- device helpers ----------------
__device__ __forceinline__ float softplus_f(float x) {
    // jax.nn.softplus == max(x,0) + log1p(exp(-|x|))
    return fmaxf(x, 0.0f) + log1pf(expf(-fabsf(x)));
}
__device__ __forceinline__ unsigned short f2bf(float f) {
    unsigned u = __float_as_uint(f);
    unsigned r = u + 0x7FFFu + ((u >> 16) & 1u);   // round-to-nearest-even
    return (unsigned short)(r >> 16);
}
__device__ __forceinline__ unsigned pack2bf(float lo, float hi) {
    return (unsigned)f2bf(lo) | ((unsigned)f2bf(hi) << 16);
}
// order-preserving float<->uint for atomic max
__device__ __forceinline__ unsigned encf(float f) {
    unsigned u = __float_as_uint(f);
    return (u & 0x80000000u) ? ~u : (u | 0x80000000u);
}
__device__ __forceinline__ float decf(unsigned v) {
    return __uint_as_float((v & 0x80000000u) ? (v & 0x7FFFFFFFu) : ~v);
}

// ---------------- setup kernels ----------------
__global__ void k_embed(const int* __restrict__ idx, const float* __restrict__ table,
                        float* __restrict__ out, int rows) {
    int i = blockIdx.x * blockDim.x + threadIdx.x;
    if (i >= rows * DIM) return;
    int r = i >> 7, c = i & 127;
    out[i] = softplus_f(table[idx[r] * DIM + c]);
}
__global__ void k_fillf(float* p, float v, int n) {
    int i = blockIdx.x * blockDim.x + threadIdx.x;
    if (i < n) p[i] = v;
}
__global__ void k_fillu(unsigned* p, unsigned v, int n) {
    int i = blockIdx.x * blockDim.x + threadIdx.x;
    if (i < n) p[i] = v;
}
__global__ void k_copy(float* __restrict__ dst, const float* __restrict__ src, int n) {
    int i = blockIdx.x * blockDim.x + threadIdx.x;
    if (i < n) dst[i] = src[i];
}

// Pack a 256x128 f32 weight (rows 0..127 from W0, 128..255 from W1) into per-lane
// WMMA B-fragment order: [colTile(8)][kb(8)][lane(32)][dword(8)], bf16 pairs.
// Lane layout (16-bit B, 32x16 K-major per column): n = lane&15, K-half = lane>>4;
// dword v holds K = kb*32 + (v>>2)*16 + half*8 + (v&3)*2 (+1 in high half).
__global__ void k_pack_w(const float* __restrict__ W0, const float* __restrict__ W1,
                         unsigned* __restrict__ out) {
    int i = blockIdx.x * blockDim.x + threadIdx.x;
    if (i >= 8 * 8 * 32 * 8) return;
    int v    = i & 7;
    int lane = (i >> 3) & 31;
    int kb   = (i >> 8) & 7;
    int ct   = i >> 11;
    int n    = ct * 16 + (lane & 15);
    int half = lane >> 4;
    int k0   = kb * 32 + ((v >> 2) << 4) + (half << 3) + ((v & 3) << 1);
    const float* Ws = (k0 < 128) ? W0 : W1;
    int kk = k0 & 127;
    out[i] = pack2bf(Ws[kk * DIM + n], Ws[(kk + 1) * DIM + n]);
}

// ---------------- fused edge GEMM (WMMA bf16, persistent blocks) ----------------
// MODE 0 (bond_conv): A[e] = [ x_i | x_j - x_i ], x_i=hb[tgt[e]], x_j=hb[src[e]];
//                     epilogue: atomicMax(enc) into agg_enc[tgt[e]*128+n]
// MODE 1 (atom_conv): A[e] = [ ha[src[e]] | hb[e] ];
//                     epilogue: atomicAdd into accum[tgt[e]*128+n]
// Block = 256 threads = 8 waves; wave w owns columns [16w,16w+16) and keeps its
// B fragments (8 x v16bf = 64 VGPRs) resident across all row tiles it processes.
template <int MODE>
__global__ __launch_bounds__(256, 2)
void k_edge_gemm(const float* __restrict__ fa,   // MODE0: hb   MODE1: ha
                 const float* __restrict__ fb,   // hb
                 const int*   __restrict__ src,
                 const int*   __restrict__ tgt,
                 const unsigned* __restrict__ Wp,
                 const float* __restrict__ bias1,
                 const float* __restrict__ bias2,
                 unsigned* __restrict__ agg_enc,
                 float*    __restrict__ accum) {
    __shared__ unsigned As[16][129];   // 16 edges x 128 dwords (bf16 pairs), +1 pad
    __shared__ int      Tg[16];        // tgt[] for the tile (for epilogue scatter)
    const int t    = threadIdx.x;
    const int lane = t & 31;
    const int wave = t >> 5;        // column tile 0..7
    const int half = lane >> 4;
    const int idx  = lane & 15;     // A: M row / B,C: N column

    // hoist this wave's B fragments into registers (reused for every row tile)
    union Frag { v16bf v; unsigned u[8]; };
    Frag Bf[8];
    {
        const unsigned* bp = Wp + ((size_t)wave * 8 * 32 + lane) * 8;
#pragma unroll
        for (int kb = 0; kb < 8; kb++) {
            const uint4* q = (const uint4*)(bp + (size_t)kb * 32 * 8);
            uint4 lo = q[0], hi = q[1];
            Bf[kb].u[0] = lo.x; Bf[kb].u[1] = lo.y; Bf[kb].u[2] = lo.z; Bf[kb].u[3] = lo.w;
            Bf[kb].u[4] = hi.x; Bf[kb].u[5] = hi.y; Bf[kb].u[6] = hi.z; Bf[kb].u[7] = hi.w;
        }
    }
    const int n = wave * 16 + idx;
    float bs = bias1[n];
    if (bias2) bs += bias2[n];

    const int el = t >> 4;          // staging: edge-in-tile for this thread
    const int c0 = (t & 15) * 8;    // staging: channel start

    for (int rt = blockIdx.x; rt < NTILES; rt += gridDim.x) {
        const int e0 = rt * 16;
        { // stage A tile into LDS with f32 -> bf16 conversion at gather time
            int e = e0 + el;
            int kt = tgt[e];
            if ((t & 15) == 0) Tg[el] = kt;
            if (MODE == 0) {
                const float* pi = fb + (size_t)kt * DIM + c0;
                const float* pj = fb + (size_t)src[e] * DIM + c0;
#pragma unroll
                for (int q = 0; q < 4; q++) {
                    float i0 = pi[2 * q], i1 = pi[2 * q + 1];
                    float j0 = pj[2 * q], j1 = pj[2 * q + 1];
                    As[el][(c0 >> 1) + q]      = pack2bf(i0, i1);
                    As[el][64 + (c0 >> 1) + q] = pack2bf(j0 - i0, j1 - i1);
                }
            } else {
                const float* pa = fa + (size_t)src[e] * DIM + c0;
                const float* pb = fb + (size_t)e * DIM + c0;
                if (e + 16 < NEDGES)  // stream prefetch of next hb tile
                    __builtin_prefetch(fb + (size_t)(e + 16) * DIM + c0, 0, 1);
#pragma unroll
                for (int q = 0; q < 4; q++) {
                    As[el][(c0 >> 1) + q]      = pack2bf(pa[2 * q], pa[2 * q + 1]);
                    As[el][64 + (c0 >> 1) + q] = pack2bf(pb[2 * q], pb[2 * q + 1]);
                }
            }
        }
        __syncthreads();

        v8f acc = {0.f, 0.f, 0.f, 0.f, 0.f, 0.f, 0.f, 0.f};
#pragma unroll
        for (int kb = 0; kb < 8; kb++) {             // K = 256 = 8 x 32
            Frag A;
#pragma unroll
            for (int v2 = 0; v2 < 8; v2++) {
                int col = kb * 16 + ((v2 >> 2) << 3) + (half << 2) + (v2 & 3);
                A.u[v2] = As[idx][col];              // A frag per ISA 16-bit layout
            }
            acc = __builtin_amdgcn_wmma_f32_16x16x32_bf16(false, A.v, false, Bf[kb].v,
                                                          (short)0, acc, false, false);
        }

        // epilogue: bias + scatter-reduce straight from C fragments
#pragma unroll
        for (int r = 0; r < 8; r++) {                // C layout: M = r + 8*half
            float val = acc[r] + bs;
            int   key = Tg[r + (half << 3)];
            if (MODE == 0)
                atomicMax(&agg_enc[(size_t)key * DIM + n], encf(val));
            else
                atomicAdd(&accum[(size_t)key * DIM + n], val);
        }
        __syncthreads();   // protect As/Tg before next tile restages
    }
}

// ---------------- pass epilogues ----------------
__global__ void k_bond_fin(const unsigned* __restrict__ agg, float* __restrict__ hb, int n) {
    int i = blockIdx.x * blockDim.x + threadIdx.x;
    if (i >= n) return;
    unsigned u = agg[i];
    float v = (u == ENC_NEGINF) ? 0.0f : decf(u);   // PyG: empty segment -> 0
    hb[i] = softplus_f(v);
}
__global__ void k_atom_fin(const float* __restrict__ acc, float* __restrict__ ha, int n) {
    int i = blockIdx.x * blockDim.x + threadIdx.x;
    if (i < n) ha[i] = softplus_f(acc[i]);
}
__global__ void k_pool(const float* __restrict__ ha, const int* __restrict__ batch,
                       float* __restrict__ pooled) {
    int i = blockIdx.x * blockDim.x + threadIdx.x;
    if (i >= NNODES * DIM) return;
    int r = i >> 7, c = i & 127;
    atomicAdd(&pooled[(size_t)batch[r] * DIM + c], ha[i]);
}

// ---------------- readout MLP (tiny) ----------------
__global__ void k_mlp0(const float* __restrict__ pooled, const float* __restrict__ W,
                       const float* __restrict__ b, float* __restrict__ h0) {
    int g = blockIdx.x, r = threadIdx.x;     // 512 x 64
    float s = b[r];
    for (int k = 0; k < DIM; k++) s += pooled[g * DIM + k] * W[k * RDIM + r];
    h0[g * RDIM + r] = softplus_f(s);
}
__global__ void k_mlp1(const float* __restrict__ h0, const float* __restrict__ W,
                       const float* __restrict__ b, float* __restrict__ h1) {
    int g = blockIdx.x, r = threadIdx.x;
    float s = b[r];
    for (int k = 0; k < RDIM; k++) s += h0[g * RDIM + k] * W[k * RDIM + r];
    h1[g * RDIM + r] = softplus_f(s);
}
__global__ void k_mlp2(const float* __restrict__ h1, const float* __restrict__ W,
                       const float* __restrict__ b, float* __restrict__ out) {
    int g = blockIdx.x * blockDim.x + threadIdx.x;
    if (g >= NGR) return;
    float s = b[0];
    for (int k = 0; k < RDIM; k++) s += h1[g * RDIM + k] * W[k];
    out[g] = s;
}

// ---------------- host launcher ----------------
extern "C" void kernel_launch(void* const* d_in, const int* in_sizes, int n_in,
                              void* d_out, int out_size, void* d_ws, size_t ws_size,
                              hipStream_t stream) {
    const int*   x        = (const int*)d_in[0];
    const int*   eattr    = (const int*)d_in[1];
    const int*   eindex   = (const int*)d_in[2];
    const int*   batch    = (const int*)d_in[3];
    const float* emb_atom = (const float*)d_in[4];
    const float* emb_bond = (const float*)d_in[5];
    const float* W_bond   = (const float*)d_in[6];
    const float* b_bond   = (const float*)d_in[7];
    const float* W_msg    = (const float*)d_in[8];
    const float* b_msg    = (const float*)d_in[9];
    const float* W_edge   = (const float*)d_in[10];
    const float* b_edge   = (const float*)d_in[11];
    const float* W_r0     = (const float*)d_in[12];
    const float* b_r0     = (const float*)d_in[13];
    const float* W_r1     = (const float*)d_in[14];
    const float* b_r1     = (const float*)d_in[15];
    const float* W_r2     = (const float*)d_in[16];
    const float* b_r2     = (const float*)d_in[17];

    const int* src = eindex;            // edge_index[0]
    const int* tgt = eindex + NEDGES;   // edge_index[1]

    // outputs live directly in d_out: [out(512) | h_atom(N*128) | h_bond(E*128)]
    float* out = (float*)d_out;
    float* ha  = out + NGR;
    float* hb  = ha + (size_t)NNODES * DIM;

    // workspace carve-out (~26.3 MB)
    char* w = (char*)d_ws;
    unsigned* aggacc = (unsigned*)w; w += (size_t)NNODES * DIM * 4;  // max-enc / sum accum
    unsigned* WbP    = (unsigned*)w; w += 16384 * 4;                 // packed W_bond
    unsigned* WcP    = (unsigned*)w; w += 16384 * 4;                 // packed [W_msg;W_edge]
    float*    pooled = (float*)w;    w += (size_t)NGR * DIM * 4;
    float*    h0     = (float*)w;    w += NGR * RDIM * 4;
    float*    h1     = (float*)w;    w += NGR * RDIM * 4;

    const int nND = NNODES * DIM;
    const int TB  = 256;

    // one-time: pack weights to bf16 fragment order; embed; constant tail of h_bond
    k_pack_w<<<64, TB, 0, stream>>>(W_bond, W_bond + 128 * DIM, WbP);
    k_pack_w<<<64, TB, 0, stream>>>(W_msg, W_edge, WcP);
    k_embed<<<(nND + TB - 1) / TB, TB, 0, stream>>>(x, emb_atom, ha, NNODES);
    k_embed<<<(nND + TB - 1) / TB, TB, 0, stream>>>(eattr, emb_bond, hb, NNODES);
    {   // rows [N,E) of h_bond are softplus(0)=ln2 after every pass (tgt<N); fill once
        int tail = (NEDGES - NNODES) * DIM;
        k_fillf<<<(tail + TB - 1) / TB, TB, 0, stream>>>(hb + (size_t)NNODES * DIM, LN2F, tail);
    }

    for (int p = 0; p < 2; p++) {
        // bond_conv: WMMA GEMM -> atomic segment-max -> softplus
        k_fillu<<<(nND + TB - 1) / TB, TB, 0, stream>>>(aggacc, ENC_NEGINF, nND);
        k_edge_gemm<0><<<GEMM_BLOCKS, TB, 0, stream>>>(hb, hb, src, tgt, WbP,
                                                       b_bond, nullptr, aggacc, nullptr);
        k_bond_fin<<<(nND + TB - 1) / TB, TB, 0, stream>>>(aggacc, hb, nND);

        // atom_conv: accum = h_atom; WMMA GEMM -> atomic segment-sum -> softplus
        k_copy<<<(nND + TB - 1) / TB, TB, 0, stream>>>((float*)aggacc, ha, nND);
        k_edge_gemm<1><<<GEMM_BLOCKS, TB, 0, stream>>>(ha, hb, src, tgt, WcP,
                                                       b_msg, b_edge, nullptr, (float*)aggacc);
        k_atom_fin<<<(nND + TB - 1) / TB, TB, 0, stream>>>((const float*)aggacc, ha, nND);
    }

    // global_add_pool + readout MLP
    k_fillf<<<(NGR * DIM + TB - 1) / TB, TB, 0, stream>>>(pooled, 0.0f, NGR * DIM);
    k_pool<<<(nND + TB - 1) / TB, TB, 0, stream>>>(ha, batch, pooled);
    k_mlp0<<<NGR, RDIM, 0, stream>>>(pooled, W_r0, b_r0, h0);
    k_mlp1<<<NGR, RDIM, 0, stream>>>(h0, W_r1, b_r1, h1);
    k_mlp2<<<2, TB, 0, stream>>>(h1, W_r2, b_r2, out);
}
